// FEMBAPatchEmbedderTest_79182017069521
// MI455X (gfx1250) — compile-verified, matrix operand order checked by reading, not run on hardware
//
#include <hip/hip_runtime.h>
#include <hip/hip_bf16.h>

// ---------------- model constants ----------------
#define BB      8
#define LL      2048          // GW
#define DMm     256           // GH*E_EMB
#define DIi     256
#define TWO_DI  512
#define NST     16
#define RR      16
#define NBLK    2
#define NCLS    10
#define HH      16
#define WW      4096
#define MROWS   (BB*LL)       // 16384
#define NCHUNK  16
#define CHUNK   (LL/NCHUNK)   // 128

typedef __attribute__((ext_vector_type(16))) _Float16 v16h;
typedef __attribute__((ext_vector_type(8)))  _Float16 v8h;
typedef __attribute__((ext_vector_type(8)))  float    v8f;

// ---------------- WMMA fragment loader ----------------
// CDNA5 16-bit operand layout (ISA 7.12.2): lane lo-half holds K = kbase+0..7
// and kbase+16..23; lane hi-half holds K = kbase+8..15 and kbase+24..31.
__device__ __forceinline__ v16h load_frag(const _Float16* __restrict__ rowp, int kbase, int hi) {
  v8h a = *(const v8h*)(rowp + kbase + 8*hi);
  v8h b = *(const v8h*)(rowp + kbase + 16 + 8*hi);
  v16h r;
#pragma unroll
  for (int i = 0; i < 8; ++i) { r[i] = a[i]; r[8+i] = b[i]; }
  return r;
}

// ---------------- register-blocked NT GEMM ----------------
// C[M,N] = A[M,K] * BT[N,K]^T  (f16 in, f32 accum/out)
// Each wave owns an MT x NT grid of 16x16 WMMA tiles: MT*NT WMMAs per k-step
// against MT+NT fragment loads -> 4x the arithmetic intensity of 1x1 tiling.
#define GEMM_WAVES 8
template <int MT, int NT>
__global__ __launch_bounds__(256) void gemm_nt_wmma(const _Float16* __restrict__ A,
                                                    const _Float16* __restrict__ BT,
                                                    float* __restrict__ C,
                                                    int M, int N, int K) {
  int lane = threadIdx.x & 31;
  int wave = threadIdx.x >> 5;
  int mblk = (blockIdx.x * GEMM_WAVES + wave) * MT;  // units of 16-row tiles
  int nblk = blockIdx.y * NT;
  if (mblk * 16 >= M) return;                        // wave-uniform; EXEC all-1s
  int hi = lane >> 4;
  int lrow = lane & 15;

  const _Float16* arow[MT];
  const _Float16* brow[NT];
#pragma unroll
  for (int i = 0; i < MT; ++i) arow[i] = A + (size_t)((mblk + i) * 16 + lrow) * K;
#pragma unroll
  for (int j = 0; j < NT; ++j) brow[j] = BT + (size_t)((nblk + j) * 16 + lrow) * K;

  v8f acc[MT][NT];
#pragma unroll
  for (int i = 0; i < MT; ++i)
#pragma unroll
    for (int j = 0; j < NT; ++j) acc[i][j] = {};

  for (int k = 0; k < K; k += 32) {
    if (k + 32 < K) {                                // global_prefetch_b8 path
#pragma unroll
      for (int i = 0; i < MT; ++i) __builtin_prefetch(arow[i] + k + 32, 0, 3);
#pragma unroll
      for (int j = 0; j < NT; ++j) __builtin_prefetch(brow[j] + k + 32, 0, 3);
    }
    v16h af[MT], bf[NT];
#pragma unroll
    for (int i = 0; i < MT; ++i) af[i] = load_frag(arow[i], k, hi);
#pragma unroll
    for (int j = 0; j < NT; ++j) bf[j] = load_frag(brow[j], k, hi);
#pragma unroll
    for (int i = 0; i < MT; ++i)
#pragma unroll
      for (int j = 0; j < NT; ++j)
        acc[i][j] = __builtin_amdgcn_wmma_f32_16x16x32_f16(
            false, af[i], false, bf[j], (short)0, acc[i][j], false, false);
  }

  // C/D layout: VGPR v -> M = 16*tile + v + 8*hi, N = 16*tile + (lane&15)
#pragma unroll
  for (int i = 0; i < MT; ++i) {
#pragma unroll
    for (int j = 0; j < NT; ++j) {
      int nr = (nblk + j) * 16 + lrow;
#pragma unroll
      for (int v = 0; v < 8; ++v) {
        int m = (mblk + i) * 16 + v + 8 * hi;
        C[(size_t)m * N + nr] = acc[i][j][v];
      }
    }
  }
}

// ---------------- weight prep: W[K,N] f32 -> WT[N,K] f16 ----------------
__global__ void transpose_to_f16(const float* __restrict__ W, _Float16* __restrict__ WT,
                                 int K, int N) {
  int idx = blockIdx.x * 256 + threadIdx.x;
  if (idx >= K * N) return;
  int n = idx / K, k = idx % K;
  WT[idx] = (_Float16)W[(size_t)k * N + n];
}

// ---------------- patch embed: x(8,1,16,4096) -> seq(8,2048,256) ----------------
__global__ void patch_embed(const float* __restrict__ x, const float* __restrict__ pw,
                            const float* __restrict__ pb, float* __restrict__ seq) {
  int idx = blockIdx.x * 256 + threadIdx.x;        // (b, w, d) d fastest
  int d = idx & 255;
  int w = (idx >> 8) & (LL - 1);
  int b = idx >> 19;
  int gh = d >> 5, e = d & 31;
  const float* xb = x + ((size_t)b * HH + gh * 2) * WW + w * 2;  // CIN==1
  const float* wp = pw + e * 4;
  seq[idx] = pb[e] + xb[0]*wp[0] + xb[1]*wp[1] + xb[WW]*wp[2] + xb[WW+1]*wp[3];
}

// ---------------- seq f32 -> f16 (fwd) + reversed f16 ----------------
__global__ void cvt_seq(const float* __restrict__ s, _Float16* __restrict__ sf,
                        _Float16* __restrict__ sr) {
  int idx = blockIdx.x * 256 + threadIdx.x;
  float v = s[idx];
  sf[idx] = (_Float16)v;
  int d = idx & 255, t = (idx >> 8) & (LL - 1), b = idx >> 19;
  sr[(((size_t)b * LL) + (LL - 1 - t)) * DMm + d] = (_Float16)v;
}

// ---------------- causal depthwise conv (K=4) + SiLU ----------------
__global__ void dwconv_silu(const float* __restrict__ xz, const float* __restrict__ cw,
                            const float* __restrict__ cb, float* __restrict__ xcf,
                            _Float16* __restrict__ xch) {
  int idx = blockIdx.x * 256 + threadIdx.x;        // (b, t, c)
  int c = idx & 255, t = (idx >> 8) & (LL - 1), b = idx >> 19;
  const float* w = cw + c * 4;
  float acc = cb[c];
#pragma unroll
  for (int k = 0; k < 4; ++k) {
    int tt = t - 3 + k;
    if (tt >= 0) acc += xz[(((size_t)b * LL + tt) * TWO_DI) + c] * w[k];
  }
  float s = acc / (1.f + __expf(-acc));            // SiLU
  xcf[idx] = s;
  xch[idx] = (_Float16)s;
}

// ---------------- dt = softplus(dt_raw @ W_dt + b_dt) ----------------
__global__ void dt_softplus(const float* __restrict__ dbl, const float* __restrict__ wdt,
                            const float* __restrict__ bdt, float* __restrict__ dt) {
  int idx = blockIdx.x * 256 + threadIdx.x;        // (m, e)
  int e = idx & 255, m = idx >> 8;
  const float* dr = dbl + (size_t)m * 48;
  float acc = bdt[e];
#pragma unroll
  for (int r = 0; r < RR; ++r) acc += dr[r] * wdt[r * DIi + e];
  dt[idx] = (acc > 20.f) ? acc : log1pf(__expf(acc));
}

// ---------------- chunked selective scan, phase 1: per-chunk partials ----------------
__global__ __launch_bounds__(256) void scan_partial(const float* __restrict__ xc,
                                                    const float* __restrict__ dtb,
                                                    const float* __restrict__ dbl,
                                                    const float* __restrict__ alog,
                                                    float* __restrict__ hpart,
                                                    float* __restrict__ pprod) {
  int b = blockIdx.x, c = blockIdx.y, e = threadIdx.x;
  float A[NST], h[NST], p[NST];
#pragma unroll
  for (int n = 0; n < NST; ++n) { A[n] = -__expf(alog[e * NST + n]); h[n] = 0.f; p[n] = 1.f; }
  __shared__ float sB[NST];
  int t0 = c * CHUNK;
  for (int t = t0; t < t0 + CHUNK; ++t) {
    size_t row = (size_t)b * LL + t;
    __syncthreads();
    if (threadIdx.x < NST) sB[threadIdx.x] = dbl[row * 48 + RR + threadIdx.x];
    __syncthreads();
    float dt = dtb[row * DIi + e];
    float dx = dt * xc[row * DIi + e];
#pragma unroll
    for (int n = 0; n < NST; ++n) {
      float da = __expf(dt * A[n]);
      h[n] = da * h[n] + dx * sB[n];
      p[n] *= da;
    }
  }
  size_t base = (((size_t)b * NCHUNK + c) * DIi + e) * NST;
#pragma unroll
  for (int n = 0; n < NST; ++n) { hpart[base + n] = h[n]; pprod[base + n] = p[n]; }
}

// ---------------- phase 2: serial prefix over chunk carries (per b,e,n) ----------------
__global__ void scan_prefix(const float* __restrict__ hpart, const float* __restrict__ pprod,
                            float* __restrict__ hin) {
  int idx = blockIdx.x * 256 + threadIdx.x;        // B*DI = 2048
  int b = idx >> 8, e = idx & 255;
  float h[NST];
#pragma unroll
  for (int n = 0; n < NST; ++n) h[n] = 0.f;
  for (int c = 0; c < NCHUNK; ++c) {
    size_t base = (((size_t)b * NCHUNK + c) * DIi + e) * NST;
#pragma unroll
    for (int n = 0; n < NST; ++n) {
      hin[base + n] = h[n];
      h[n] = pprod[base + n] * h[n] + hpart[base + n];
    }
  }
}

// ---------------- phase 3: rescan with carries, produce yz = y * silu(z) (f16) ----------------
__global__ __launch_bounds__(256) void scan_final(const float* __restrict__ xc,
                                                  const float* __restrict__ dtb,
                                                  const float* __restrict__ dbl,
                                                  const float* __restrict__ xz,
                                                  const float* __restrict__ alog,
                                                  const float* __restrict__ dsk,
                                                  const float* __restrict__ hin,
                                                  _Float16* __restrict__ yz) {
  int b = blockIdx.x, c = blockIdx.y, e = threadIdx.x;
  float A[NST], h[NST];
  size_t hbase = (((size_t)b * NCHUNK + c) * DIi + e) * NST;
#pragma unroll
  for (int n = 0; n < NST; ++n) { A[n] = -__expf(alog[e * NST + n]); h[n] = hin[hbase + n]; }
  float D = dsk[e];
  __shared__ float sB[NST], sC[NST];
  int t0 = c * CHUNK;
  for (int t = t0; t < t0 + CHUNK; ++t) {
    size_t row = (size_t)b * LL + t;
    __syncthreads();
    if (threadIdx.x < 2 * NST) {
      float v = dbl[row * 48 + RR + threadIdx.x];
      if (threadIdx.x < NST) sB[threadIdx.x] = v; else sC[threadIdx.x - NST] = v;
    }
    __syncthreads();
    float dt = dtb[row * DIi + e];
    float x  = xc[row * DIi + e];
    float dx = dt * x;
    float y  = x * D;
#pragma unroll
    for (int n = 0; n < NST; ++n) {
      float da = __expf(dt * A[n]);
      h[n] = da * h[n] + dx * sB[n];
      y += h[n] * sC[n];
    }
    float z  = xz[row * TWO_DI + DIi + e];
    float sz = z / (1.f + __expf(-z));
    yz[row * DIi + e] = (_Float16)(y * sz);
  }
}

// ---------------- combine fwd + reversed rev into next seq ----------------
__global__ void combine_dirs(const float* __restrict__ y0, const float* __restrict__ y1,
                             float* __restrict__ s) {
  int idx = blockIdx.x * 256 + threadIdx.x;
  int d = idx & 255, t = (idx >> 8) & (LL - 1), b = idx >> 19;
  s[idx] = y0[idx] + y1[(((size_t)b * LL) + (LL - 1 - t)) * DMm + d];
}

// ---------------- mean pool over L ----------------
__global__ void pool_mean(const float* __restrict__ s, float* __restrict__ pooled) {
  int b = blockIdx.x, d = threadIdx.x;
  float acc = 0.f;
  for (int t = 0; t < LL; ++t) acc += s[((size_t)b * LL + t) * DMm + d];
  pooled[b * DMm + d] = acc * (1.f / (float)LL);
}

// ---------------- classifier: pooled @ cls_w.T + cls_b ----------------
__global__ void classify(const float* __restrict__ pooled, const float* __restrict__ cw,
                         const float* __restrict__ cb, float* __restrict__ out) {
  int tid = blockIdx.x * blockDim.x + threadIdx.x;
  if (tid >= BB * NCLS) return;
  int b = tid / NCLS, c = tid % NCLS;
  float acc = cb[c];
  for (int d = 0; d < DMm; ++d) acc += pooled[b * DMm + d] * cw[c * DMm + d];
  out[b * NCLS + c] = acc;
}

// ---------------- host orchestration ----------------
extern "C" void kernel_launch(void* const* d_in, const int* in_sizes, int n_in,
                              void* d_out, int out_size, void* d_ws, size_t ws_size,
                              hipStream_t stream) {
  const float* x    = (const float*)d_in[0];
  const float* pw   = (const float*)d_in[1];
  const float* pb   = (const float*)d_in[2];
  const float* Win  = (const float*)d_in[3];   // (NB,2,DM,2DI)
  const float* cw   = (const float*)d_in[4];   // (NB,2,DI,1,K)
  const float* cb   = (const float*)d_in[5];   // (NB,2,DI)
  const float* Wx   = (const float*)d_in[6];   // (NB,2,DI,48)
  const float* Wdt  = (const float*)d_in[7];   // (NB,2,R,DI)
  const float* bdt  = (const float*)d_in[8];   // (NB,2,DI)
  const float* Alog = (const float*)d_in[9];   // (NB,2,DI,N)
  const float* Dsk  = (const float*)d_in[10];  // (NB,2,DI)
  const float* Wout = (const float*)d_in[11];  // (NB,2,DI,DM)
  const float* clsw = (const float*)d_in[12];
  const float* clsb = (const float*)d_in[13];
  float* out = (float*)d_out;

  // --- carve workspace ---
  char* base = (char*)d_ws;
  size_t off = 0;
  auto take = [&](size_t bytes) -> char* {
    char* p = base + off;
    off += (bytes + 255) & ~(size_t)255;
    return p;
  };
  _Float16* wtin[4];  for (int j = 0; j < 4; ++j) wtin[j]  = (_Float16*)take((size_t)TWO_DI * DMm * 2);
  _Float16* wtx[4];   for (int j = 0; j < 4; ++j) wtx[j]   = (_Float16*)take((size_t)48 * DIi * 2);
  _Float16* wtout[4]; for (int j = 0; j < 4; ++j) wtout[j] = (_Float16*)take((size_t)DMm * DIi * 2);
  float*    seqf  = (float*)   take((size_t)MROWS * DMm * 4);
  _Float16* seqh  = (_Float16*)take((size_t)MROWS * DMm * 2);
  _Float16* seqrh = (_Float16*)take((size_t)MROWS * DMm * 2);
  float*    xz    = (float*)   take((size_t)MROWS * TWO_DI * 4);
  float*    xcf   = (float*)   take((size_t)MROWS * DIi * 4);
  _Float16* xch   = (_Float16*)take((size_t)MROWS * DIi * 2);
  float*    dblb  = (float*)   take((size_t)MROWS * 48 * 4);
  float*    dtb   = (float*)   take((size_t)MROWS * DIi * 4);
  _Float16* yzh   = (_Float16*)take((size_t)MROWS * DIi * 2);
  float*    yd0   = (float*)   take((size_t)MROWS * DMm * 4);
  float*    yd1   = (float*)   take((size_t)MROWS * DMm * 4);
  float*    hpart = (float*)   take((size_t)BB * NCHUNK * DIi * NST * 4);
  float*    pprod = (float*)   take((size_t)BB * NCHUNK * DIi * NST * 4);
  float*    hinb  = (float*)   take((size_t)BB * NCHUNK * DIi * NST * 4);
  float*    pooled= (float*)   take((size_t)BB * DMm * 4);
  if (off > ws_size) return;   // workspace too small; nothing safe to do

  const int ELT = MROWS * DMm;          // 4,194,304
  const int ELT_BLKS = ELT / 256;       // 16384

  // --- weight prep (f32 -> f16 transposed) ---
  for (int j = 0; j < 4; ++j) {
    transpose_to_f16<<<(DMm * TWO_DI + 255) / 256, 256, 0, stream>>>(
        Win + (size_t)j * DMm * TWO_DI, wtin[j], DMm, TWO_DI);
    transpose_to_f16<<<(DIi * 48 + 255) / 256, 256, 0, stream>>>(
        Wx + (size_t)j * DIi * 48, wtx[j], DIi, 48);
    transpose_to_f16<<<(DIi * DMm + 255) / 256, 256, 0, stream>>>(
        Wout + (size_t)j * DIi * DMm, wtout[j], DIi, DMm);
  }

  // --- patch embed ---
  patch_embed<<<ELT_BLKS, 256, 0, stream>>>(x, pw, pb, seqf);

  // M-tiles = 1024; each block = 8 waves x MT(=4) tiles = 32 M-tiles
  const int GRIDM4 = MROWS / (16 * GEMM_WAVES * 4);   // 32
  for (int i = 0; i < NBLK; ++i) {
    cvt_seq<<<ELT_BLKS, 256, 0, stream>>>(seqf, seqh, seqrh);
    for (int dir = 0; dir < 2; ++dir) {
      int j = i * 2 + dir;
      const _Float16* aseq = dir ? seqrh : seqh;
      // xz = seq @ W_in   (M x 512), 4x4 register-blocked WMMA
      gemm_nt_wmma<4, 4><<<dim3(GRIDM4, TWO_DI / 64), 256, 0, stream>>>(
          aseq, wtin[j], xz, MROWS, TWO_DI, DMm);
      // depthwise causal conv + SiLU
      dwconv_silu<<<ELT_BLKS, 256, 0, stream>>>(
          xz, cw + (size_t)j * DIi * 4, cb + (size_t)j * DIi, xcf, xch);
      // dbl = xc @ W_x   (M x 48), 4x3 register-blocked WMMA (N=48 exact)
      gemm_nt_wmma<4, 3><<<dim3(GRIDM4, 1), 256, 0, stream>>>(
          xch, wtx[j], dblb, MROWS, 48, DIi);
      // dt = softplus(dt_raw @ W_dt + b_dt)
      dt_softplus<<<ELT_BLKS, 256, 0, stream>>>(
          dblb, Wdt + (size_t)j * RR * DIi, bdt + (size_t)j * DIi, dtb);
      // chunk-parallel selective scan
      scan_partial<<<dim3(BB, NCHUNK), 256, 0, stream>>>(
          xcf, dtb, dblb, Alog + (size_t)j * DIi * NST, hpart, pprod);
      scan_prefix<<<BB, 256, 0, stream>>>(hpart, pprod, hinb);
      scan_final<<<dim3(BB, NCHUNK), 256, 0, stream>>>(
          xcf, dtb, dblb, xz, Alog + (size_t)j * DIi * NST,
          Dsk + (size_t)j * DIi, hinb, yzh);
      // y_out = (y * silu(z)) @ W_out   (M x 256), 4x4 register-blocked
      gemm_nt_wmma<4, 4><<<dim3(GRIDM4, DMm / 64), 256, 0, stream>>>(
          yzh, wtout[j], dir ? yd1 : yd0, MROWS, DMm, DIi);
    }
    combine_dirs<<<ELT_BLKS, 256, 0, stream>>>(yd0, yd1, seqf);
  }

  pool_mean<<<BB, DMm, 0, stream>>>(seqf, pooled);
  classify<<<1, 128, 0, stream>>>(pooled, clsw, clsb, out);
}
